// EncMtanRnn_12747462934581
// MI455X (gfx1250) — compile-verified, matrix operand order checked by reading, not run on hardware
//
#include <hip/hip_runtime.h>
#include <hip/hip_bf16.h>

// ---------------------------------------------------------------------------
// mTAN encoder for MI455X (gfx1250, wave32, WMMA + async global->LDS).
// Dims: B=4, LK=256, LQ=128, D=32 (2D=64), ET=128, H=4, DK=32, NH=128, LAT=20.
// All dense reductions go through v_wmma_f32_16x16x32_f16 (f16 operands,
// f32 accumulate), fed from LDS tiles staged with GLOBAL_LOAD_ASYNC_TO_LDS_B128
// (ASYNCcnt + s_wait_asynccnt). Masked per-feature softmax is re-expressed as
// two GEMMs (numer/denom) against exp(scores - rowmax), which is exact.
// ---------------------------------------------------------------------------

typedef __attribute__((ext_vector_type(16))) _Float16 v16h;
typedef __attribute__((ext_vector_type(8)))  float    v8f;

#define C_B   4
#define C_LK  256
#define C_LQ  128
#define C_D   32
#define C_2D  64
#define C_ET  128
#define C_H   4
#define C_DK  32
#define C_NH  128

// ---------------------------------------------------------------------------
// Async global->LDS copy (CDNA5). LDS dest offset comes from the low 32 bits
// of the generic shared pointer (ISA 10.2).
// Tracked by ASYNCcnt; waited with s_wait_asynccnt.
// ---------------------------------------------------------------------------
__device__ __forceinline__ unsigned to_lds(const void* p) {
  return (unsigned)(unsigned long long)p;
}
__device__ __forceinline__ void async_g2l_b128(unsigned ldsoff, const _Float16* g) {
  asm volatile("global_load_async_to_lds_b128 %0, %1, off"
               :: "v"(ldsoff), "v"(g)
               : "memory");
}
__device__ __forceinline__ void wait_async0() {
  asm volatile("s_wait_asynccnt 0" ::: "memory");
}

// ---------------------------------------------------------------------------
// Fragment loader per CDNA5 ISA §7.12.2 (wave32).
// A (16x32, MxK, f16): lane L: row = L&15, kbase = (L>>4)*8,
//   halves 0..7 -> K = kbase+0..7, halves 8..15 -> K = 16+kbase+0..7.
// B (32x16, KxN, f16) mirrors A with N in place of M; since we compute
//   C = A @ W^T with W stored (N x K) row-major, the B fragment for column n
//   is simply row n of W -- contiguous along K, identical addressing to A.
// ---------------------------------------------------------------------------
__device__ __forceinline__ v16h load_frag_rowK(const _Float16* base, int ld) {
  const int lane = threadIdx.x;            // 0..31
  const int r    = lane & 15;              // row (A) or column (B)
  const int kb   = (lane >> 4) * 8;
  const _Float16* p = base + (size_t)r * ld;
  v16h f;
#pragma unroll
  for (int i = 0; i < 8; ++i) {
    const int k0 = (i < 4) ? (kb + 2 * i) : (16 + kb + 2 * (i - 4));
    f[2 * i]     = p[k0];
    f[2 * i + 1] = p[k0 + 1];
  }
  return f;
}

// ---------------------------------------------------------------------------
// Generic batched GEMM: C[batch] = A[batch] @ W[batch]^T * scale + bias.
//   A: (M x K) f16 row-major, leading dim lda, batch offset b*sAb + h*sAh
//   W: (N x K) f16 row-major, leading dim ldw, batch offset b*sWb + h*sWh
//   C: (M x N), leading dim ldc, batch offset batch*sC; f32 and/or f16 stores.
// block = (32,4): 4 waves; block tile = 64(M) x 16(N). Per K-chunk the A
// (64x32) and W (16x32) f16 tiles are staged to LDS via async copies, then
// each wave runs one 16x16x32 WMMA from LDS. grid = (N/16, M/64, nb).
// No bounds checks: all dims divide exactly -> EXEC all-ones for WMMA.
// ---------------------------------------------------------------------------
__global__ void gemm_wt_f16(const _Float16* __restrict__ A, int lda, long sAb, long sAh,
                            const _Float16* __restrict__ W, int ldw, long sWb, long sWh,
                            float* __restrict__ Cf, _Float16* __restrict__ Ch,
                            int ldc, long sC, int K,
                            const float* __restrict__ bias, float scale, int Hdiv) {
  __shared__ _Float16 shA[64 * 32];
  __shared__ _Float16 shW[16 * 32];

  const int batch = blockIdx.z;
  const int b = batch / Hdiv;
  const int h = batch % Hdiv;
  A += (size_t)b * sAb + (size_t)h * sAh;
  W += (size_t)b * sWb + (size_t)h * sWh;
  const size_t coff = (size_t)batch * sC;

  const int tileN  = blockIdx.x;
  const int mblock = blockIdx.y;                       // 64 rows of M
  const int tid    = threadIdx.y * 32 + threadIdx.x;   // 0..127

  const _Float16* Abase = A + (size_t)mblock * 64 * lda;
  const _Float16* Wbase = W + (size_t)tileN * 16 * ldw;

  v8f acc = {};
  for (int kc = 0; kc < K; kc += 32) {
    // stage A tile: 64 rows x 32 halfs = 256 x 16B, 2 transfers per thread
#pragma unroll
    for (int t = 0; t < 2; ++t) {
      const int tr  = tid * 2 + t;
      const int row = tr >> 2, seg = tr & 3;
      async_g2l_b128(to_lds(&shA[row * 32 + seg * 8]),
                     Abase + (size_t)row * lda + kc + seg * 8);
    }
    // stage W tile: 16 rows x 32 halfs = 64 x 16B
    if (tid < 64) {
      const int row = tid >> 2, seg = tid & 3;
      async_g2l_b128(to_lds(&shW[row * 32 + seg * 8]),
                     Wbase + (size_t)row * ldw + kc + seg * 8);
    }
    wait_async0();        // this wave's async transfers complete
    __syncthreads();      // all waves' staged data visible

    v16h af = load_frag_rowK(shA + threadIdx.y * 16 * 32, 32);
    v16h bf = load_frag_rowK(shW, 32);
    acc = __builtin_amdgcn_wmma_f32_16x16x32_f16(
        /*neg_a=*/false, af, /*neg_b=*/false, bf,
        /*c_mod=*/(short)0, acc, /*reuse_a=*/false, /*reuse_b=*/false);

    __syncthreads();      // protect LDS tiles before next chunk overwrites
  }

  const int lane  = threadIdx.x;
  const int ncol  = lane & 15;
  const int mbase = (lane >> 4) * 8;
  const int gcol  = tileN * 16 + ncol;
  const int tileM = mblock * 4 + threadIdx.y;
  const float bval = bias ? bias[gcol] : 0.0f;
#pragma unroll
  for (int r = 0; r < 8; ++r) {
    const int grow = tileM * 16 + mbase + r;
    const float v = acc[r] * scale + bval;
    const size_t off = coff + (size_t)grow * ldc + gcol;
    if (Cf) Cf[off] = v;
    if (Ch) Ch[off] = (_Float16)v;
  }
}

// ---------------------------------------------------------------------------
// Small helper kernels
// ---------------------------------------------------------------------------
__global__ void cast_f32_f16(const float* __restrict__ a, _Float16* __restrict__ o, int n) {
  int i = blockIdx.x * 256 + threadIdx.x;
  if (i < n) o[i] = (_Float16)a[i];
}

// time embedding: out[row][0] = t*w_lin + b_lin ; out[row][j] = sin(t*w_per[j-1]+b_per[j-1])
__global__ void embed_kernel(const float* __restrict__ t,
                             const float* __restrict__ w_lin, const float* __restrict__ b_lin,
                             const float* __restrict__ w_per, const float* __restrict__ b_per,
                             _Float16* __restrict__ out) {
  const int row = blockIdx.x;
  const int c = threadIdx.x;                 // 0..127
  const float tv = t[row];
  float v;
  if (c == 0) v = tv * w_lin[0] + b_lin[0];
  else        v = sinf(tv * w_per[c - 1] + b_per[c - 1]);
  out[(size_t)row * C_ET + c] = (_Float16)v;
}

// stage x into transposed f16 operands for the numer/denom GEMMs:
//   xmT[(b*64+d)*LK + k] = x[b,k,d] * mask[b,k,d];  mT[...] = mask[b,k,d]
// mask[d] = x[b,k, D + (d % D)]
__global__ void stage_x(const float* __restrict__ x,
                        _Float16* __restrict__ xmT, _Float16* __restrict__ mT) {
  const int bk = blockIdx.x;                 // b*LK + k
  const int d = threadIdx.x;                 // 0..63
  const int b = bk >> 8, k = bk & 255;
  const float* xr = x + (size_t)bk * C_2D;
  const float xv = xr[d];
  const float mv = xr[C_D + (d & (C_D - 1))];
  const size_t off = ((size_t)(b * C_2D + d)) * C_LK + k;
  xmT[off] = (_Float16)(xv * mv);
  mT[off]  = (_Float16)mv;
}

// per-row (length LK=256) max reduce then E = exp(s - max) in f16
__global__ void rowmax_exp(const float* __restrict__ S, _Float16* __restrict__ E) {
  __shared__ float red[8];
  const int row = blockIdx.x;                // B*H*LQ rows
  const int tid = threadIdx.x;               // 256 threads
  const float v = S[(size_t)row * C_LK + tid];
  float m = v;
#pragma unroll
  for (int o = 16; o > 0; o >>= 1) m = fmaxf(m, __shfl_xor(m, o, 32));
  if ((tid & 31) == 0) red[tid >> 5] = m;
  __syncthreads();
  if (tid == 0) {
    float t = red[0];
#pragma unroll
    for (int i = 1; i < 8; ++i) t = fmaxf(t, red[i]);
    red[0] = t;
  }
  __syncthreads();
  const float mx = red[0];
  E[(size_t)row * C_LK + tid] = (_Float16)expf(v - mx);
}

// att[(b*LQ+q)*256 + h*64 + d] = denom>0 ? numer/denom : 0   (f16)
// numer/denom layout: ((b*H+h)*LQ + q)*64 + d
__global__ void att_div(const float* __restrict__ numer, const float* __restrict__ denom,
                        _Float16* __restrict__ att) {
  const int idx = blockIdx.x * 256 + threadIdx.x;   // 131072 total
  const int d = idx & 63;
  const int q = (idx >> 6) & 127;
  const int h = (idx >> 13) & 3;
  const int b = idx >> 15;
  const float n = numer[idx], de = denom[idx];
  const float v = de > 0.0f ? n / de : 0.0f;
  att[((size_t)(b * C_LQ + q)) * (C_H * C_2D) + h * C_2D + d] = (_Float16)v;
}

// ---------------------------------------------------------------------------
// Bidirectional GRU, persistent: 2 blocks (dir) x 512 threads (b=tid/128,
// j=tid%128). h state (4x128) lives in LDS; whh streamed through L2 (196KB,
// resident in 192MB L2). gi = xs @ wih^T + bih precomputed by WMMA GEMM,
// layout (dir, b*LQ+q, 384). hh[(b*LQ+t)*256 + dir*128 + j] = hidden.
// ---------------------------------------------------------------------------
__global__ void gru_kernel(const float* __restrict__ gi,
                           const float* __restrict__ whh_f, const float* __restrict__ bhh_f,
                           const float* __restrict__ whh_b, const float* __restrict__ bhh_b,
                           float* __restrict__ hh) {
  __shared__ float hsh[C_B * C_NH];
  const int dir = blockIdx.x;
  const float* whh = dir ? whh_b : whh_f;
  const float* bhh = dir ? bhh_b : bhh_f;
  const float* g = gi + (size_t)dir * (C_B * C_LQ) * (3 * C_NH);
  const int tid = threadIdx.x;               // 0..511
  const int b = tid >> 7, j = tid & 127;
  hsh[tid] = 0.0f;
  __syncthreads();

  const float* wr = whh + (size_t)j * C_NH;
  const float* wz = whh + (size_t)(C_NH + j) * C_NH;
  const float* wn = whh + (size_t)(2 * C_NH + j) * C_NH;
  const float* hb = hsh + b * C_NH;

  for (int s = 0; s < C_LQ; ++s) {
    const int tq = dir ? (C_LQ - 1 - s) : s;
    const float* gr = g + ((size_t)(b * C_LQ + tq)) * (3 * C_NH);
    float ghr = bhh[j], ghz = bhh[C_NH + j], ghn = bhh[2 * C_NH + j];
    for (int kk = 0; kk < C_NH; ++kk) {
      const float hv = hb[kk];
      ghr += hv * wr[kk];
      ghz += hv * wz[kk];
      ghn += hv * wn[kk];
    }
    const float r = 1.0f / (1.0f + expf(-(gr[j] + ghr)));
    const float z = 1.0f / (1.0f + expf(-(gr[C_NH + j] + ghz)));
    const float n = tanhf(gr[2 * C_NH + j] + r * ghn);
    const float hnew = (1.0f - z) * n + z * hb[j];
    __syncthreads();
    hsh[tid] = hnew;
    __syncthreads();
    hh[((size_t)(b * C_LQ + tq)) * (2 * C_NH) + dir * C_NH + j] = hnew;
  }
}

// final MLP: hid = relu(hh_row @ w1^T + b1) (50); out = hid @ w2^T + b2 (40)
__global__ void mlp_kernel(const float* __restrict__ hh,
                           const float* __restrict__ w1, const float* __restrict__ b1,
                           const float* __restrict__ w2, const float* __restrict__ b2,
                           float* __restrict__ out) {
  __shared__ float row[2 * C_NH];
  __shared__ float hid[50];
  const int r = blockIdx.x;                  // 0..511 (b*LQ+q)
  const int tid = threadIdx.x;               // 64
  for (int i = tid; i < 2 * C_NH; i += 64) row[i] = hh[(size_t)r * (2 * C_NH) + i];
  __syncthreads();
  if (tid < 50) {
    float a = b1[tid];
    const float* w = w1 + (size_t)tid * (2 * C_NH);
    for (int k = 0; k < 2 * C_NH; ++k) a += row[k] * w[k];
    hid[tid] = fmaxf(a, 0.0f);
  }
  __syncthreads();
  if (tid < 40) {
    float a = b2[tid];
    const float* w = w2 + (size_t)tid * 50;
    for (int k = 0; k < 50; ++k) a += hid[k] * w[k];
    out[(size_t)r * 40 + tid] = a;
  }
}

// ---------------------------------------------------------------------------
// Launch
// ---------------------------------------------------------------------------
extern "C" void kernel_launch(void* const* d_in, const int* in_sizes, int n_in,
                              void* d_out, int out_size, void* d_ws, size_t ws_size,
                              hipStream_t stream) {
  const float* x      = (const float*)d_in[0];
  const float* tsteps = (const float*)d_in[1];
  const float* q_ref  = (const float*)d_in[2];
  const float* w_lin  = (const float*)d_in[3];
  const float* b_lin  = (const float*)d_in[4];
  const float* w_per  = (const float*)d_in[5];
  const float* b_per  = (const float*)d_in[6];
  const float* wq     = (const float*)d_in[7];
  const float* bq     = (const float*)d_in[8];
  const float* wk     = (const float*)d_in[9];
  const float* bk     = (const float*)d_in[10];
  const float* wo     = (const float*)d_in[11];
  const float* bo     = (const float*)d_in[12];
  const float* wih_f  = (const float*)d_in[13];
  const float* whh_f  = (const float*)d_in[14];
  const float* bih_f  = (const float*)d_in[15];
  const float* bhh_f  = (const float*)d_in[16];
  const float* wih_b  = (const float*)d_in[17];
  const float* whh_b  = (const float*)d_in[18];
  const float* bih_b  = (const float*)d_in[19];
  const float* bhh_b  = (const float*)d_in[20];
  const float* w1     = (const float*)d_in[21];
  const float* b1     = (const float*)d_in[22];
  const float* w2     = (const float*)d_in[23];
  const float* b2     = (const float*)d_in[24];
  float* out = (float*)d_out;

  // workspace carve (256B aligned)
  char* p = (char*)d_ws;
  auto alloc = [&](size_t bytes) -> char* {
    char* r = p; p += (bytes + 255) & ~(size_t)255; return r;
  };
  _Float16* key_e  = (_Float16*)alloc(C_B * C_LK * C_ET * 2);   // (B*LK, ET)
  _Float16* qry_e  = (_Float16*)alloc(C_LQ * C_ET * 2);         // (LQ, ET)
  _Float16* wq_h   = (_Float16*)alloc(C_ET * C_ET * 2);
  _Float16* wk_h   = (_Float16*)alloc(C_ET * C_ET * 2);
  _Float16* wo_h   = (_Float16*)alloc(C_NH * C_H * C_2D * 2);
  _Float16* wihf_h = (_Float16*)alloc(3 * C_NH * C_NH * 2);
  _Float16* wihb_h = (_Float16*)alloc(3 * C_NH * C_NH * 2);
  _Float16* q_h    = (_Float16*)alloc(C_LQ * C_ET * 2);         // (LQ, ET)
  _Float16* k_h    = (_Float16*)alloc(C_B * C_LK * C_ET * 2);   // (B*LK, ET)
  float*    sc_f   = (float*)   alloc((size_t)C_B * C_H * C_LQ * C_LK * 4);
  _Float16* E_h    = (_Float16*)alloc((size_t)C_B * C_H * C_LQ * C_LK * 2);
  _Float16* xmT    = (_Float16*)alloc(C_B * C_2D * C_LK * 2);
  _Float16* mT     = (_Float16*)alloc(C_B * C_2D * C_LK * 2);
  float*    numer  = (float*)   alloc((size_t)C_B * C_H * C_LQ * C_2D * 4);
  float*    denom  = (float*)   alloc((size_t)C_B * C_H * C_LQ * C_2D * 4);
  _Float16* att_h  = (_Float16*)alloc(C_B * C_LQ * C_H * C_2D * 2);  // (512, 256)
  _Float16* out_h  = (_Float16*)alloc(C_B * C_LQ * C_NH * 2);        // (512, 128)
  float*    gi_f   = (float*)   alloc((size_t)2 * C_B * C_LQ * 3 * C_NH * 4);
  float*    hh_f   = (float*)   alloc((size_t)C_B * C_LQ * 2 * C_NH * 4);

  const dim3 gblk(32, 4, 1);  // 4 waves; block tile 64(M) x 16(N)
  const float inv_sqrt_dk = 0.17677669529663687f;  // 1/sqrt(32)

  // 1) weight casts to f16
  cast_f32_f16<<<dim3(64), dim3(256), 0, stream>>>(wq, wq_h, C_ET * C_ET);
  cast_f32_f16<<<dim3(64), dim3(256), 0, stream>>>(wk, wk_h, C_ET * C_ET);
  cast_f32_f16<<<dim3(128), dim3(256), 0, stream>>>(wo, wo_h, C_NH * C_H * C_2D);
  cast_f32_f16<<<dim3(192), dim3(256), 0, stream>>>(wih_f, wihf_h, 3 * C_NH * C_NH);
  cast_f32_f16<<<dim3(192), dim3(256), 0, stream>>>(wih_b, wihb_h, 3 * C_NH * C_NH);

  // 2) time embeddings (keys: B*LK rows; queries: LQ rows)
  embed_kernel<<<dim3(C_B * C_LK), dim3(C_ET), 0, stream>>>(tsteps, w_lin, b_lin, w_per, b_per, key_e);
  embed_kernel<<<dim3(C_LQ), dim3(C_ET), 0, stream>>>(q_ref, w_lin, b_lin, w_per, b_per, qry_e);

  // 3) stage x (masked values + mask, transposed, f16)
  stage_x<<<dim3(C_B * C_LK), dim3(C_2D), 0, stream>>>(x, xmT, mT);

  // 4) q = qry_e @ wq^T + bq -> f16   (M=128,N=128,K=128)
  gemm_wt_f16<<<dim3(8, 2, 1), gblk, 0, stream>>>(
      qry_e, C_ET, 0, 0, wq_h, C_ET, 0, 0,
      nullptr, q_h, C_ET, 0, C_ET, bq, 1.0f, 1);

  // 5) k = key_e @ wk^T + bk -> f16   (M=1024,N=128,K=128)
  gemm_wt_f16<<<dim3(8, 16, 1), gblk, 0, stream>>>(
      key_e, C_ET, 0, 0, wk_h, C_ET, 0, 0,
      nullptr, k_h, C_ET, 0, C_ET, bk, 1.0f, 1);

  // 6) scores[b,h] = q[h] @ k[b,h]^T / sqrt(DK)  (M=128,N=256,K=32, batch=16)
  gemm_wt_f16<<<dim3(16, 2, C_B * C_H), gblk, 0, stream>>>(
      q_h, C_ET, 0, C_DK,
      k_h, C_ET, (long)C_LK * C_ET, C_DK,
      sc_f, nullptr, C_LK, (long)C_LQ * C_LK, C_DK,
      nullptr, inv_sqrt_dk, C_H);

  // 7) E = exp(scores - rowmax)  -> f16
  rowmax_exp<<<dim3(C_B * C_H * C_LQ), dim3(C_LK), 0, stream>>>(sc_f, E_h);

  // 8) numer[b,h] = E[b,h] @ xmT[b]^T  (M=128,N=64,K=256, batch=16)
  gemm_wt_f16<<<dim3(4, 2, C_B * C_H), gblk, 0, stream>>>(
      E_h, C_LK, (long)C_H * C_LQ * C_LK, (long)C_LQ * C_LK,
      xmT, C_LK, (long)C_2D * C_LK, 0,
      numer, nullptr, C_2D, (long)C_LQ * C_2D, C_LK,
      nullptr, 1.0f, C_H);

  // 9) denom[b,h] = E[b,h] @ mT[b]^T
  gemm_wt_f16<<<dim3(4, 2, C_B * C_H), gblk, 0, stream>>>(
      E_h, C_LK, (long)C_H * C_LQ * C_LK, (long)C_LQ * C_LK,
      mT, C_LK, (long)C_2D * C_LK, 0,
      denom, nullptr, C_2D, (long)C_LQ * C_2D, C_LK,
      nullptr, 1.0f, C_H);

  // 10) att = numer/denom, gathered to (b*LQ+q, h*64+d) f16
  att_div<<<dim3(512), dim3(256), 0, stream>>>(numer, denom, att_h);

  // 11) out = att @ wo^T + bo -> f16  (M=512,N=128,K=256)
  gemm_wt_f16<<<dim3(8, 8, 1), gblk, 0, stream>>>(
      att_h, C_H * C_2D, 0, 0, wo_h, C_H * C_2D, 0, 0,
      nullptr, out_h, C_NH, 0, C_H * C_2D, bo, 1.0f, 1);

  // 12) gi = out @ wih^T + bih, both directions (M=512,N=384,K=128)
  gemm_wt_f16<<<dim3(24, 8, 1), gblk, 0, stream>>>(
      out_h, C_NH, 0, 0, wihf_h, C_NH, 0, 0,
      gi_f, nullptr, 3 * C_NH, 0, C_NH, bih_f, 1.0f, 1);
  gemm_wt_f16<<<dim3(24, 8, 1), gblk, 0, stream>>>(
      out_h, C_NH, 0, 0, wihb_h, C_NH, 0, 0,
      gi_f + (size_t)C_B * C_LQ * 3 * C_NH, nullptr, 3 * C_NH, 0, C_NH, bih_b, 1.0f, 1);

  // 13) bidirectional GRU (persistent, sequential over LQ)
  gru_kernel<<<dim3(2), dim3(512), 0, stream>>>(gi_f, whh_f, bhh_f, whh_b, bhh_b, hh_f);

  // 14) final MLP -> d_out (B,LQ,40)
  mlp_kernel<<<dim3(C_B * C_LQ), dim3(64), 0, stream>>>(hh_f, w1, b1, w2, b2, out);
}
